// DCWF_41180146434871
// MI455X (gfx1250) — matrix operand order repaired and checked by reading, not compiled, not run on hardware
//
#include <hip/hip_runtime.h>

// ---------------------------------------------------------------------------
// Fused deformable-conv + fusion-path kernel for MI455X (gfx1250, wave32).
// All GEMMs run on v_wmma_f32_16x16x32_f16 (f16 operands, f32 accumulate).
// One workgroup = 32 output positions x all 128 output channels.
// ---------------------------------------------------------------------------

typedef __attribute__((ext_vector_type(16))) _Float16 v16h;
typedef __attribute__((ext_vector_type(8)))  _Float16 v8h;
typedef __attribute__((ext_vector_type(8)))  float    v8f;

#define KTOT    1152      // K = Cin(128) * KK(9)
#define KCH     36        // K chunks of 32
#define SPITCH  1160      // im2col LDS row pitch (halves), padded vs 1152
#define XCPITCH 136       // center-slice pitch (halves), padded vs 128
#define HPITCH  72        // fusion-mid pitch (halves), padded vs 64

// ---- WMMA fragment helpers (layouts per CDNA5 ISA 7.12.2) ------------------

// 16-bit A-matrix 16x32: lane L holds row m = L%16; element e of the v16h
// maps to k(e) = (e<8 ? 0 : 16) + 8*(L/16) + (e&7)  ->  two contiguous runs
// of 8 halves at k0 + g*8 and k0 + 16 + g*8 (row-major source).
__device__ __forceinline__ v16h load_a_frag(const _Float16* __restrict__ W,
                                            int pitch, int m0, int k0, int lane) {
  int m = m0 + (lane & 15);
  int g = lane >> 4;
  const _Float16* p = W + m * pitch + k0 + g * 8;
  v8h lo = *(const v8h*)p;
  v8h hi = *(const v8h*)(p + 16);
  return __builtin_shufflevector(lo, hi, 0,1,2,3,4,5,6,7,8,9,10,11,12,13,14,15);
}

// 16-bit B-matrix 32x16: lane L holds col n = L%16; element e maps to
// k(e) = (e<8 ? 0 : 16) + 8*(L/16) + (e&7)  ->  two runs of 8 halves at
// k0 + g*8 and k0 + 16 + g*8 in a [n][k] row-major LDS buffer.
__device__ __forceinline__ v16h load_b_frag(const _Float16* Sb,
                                            int pitch, int n0, int k0, int lane) {
  int n = n0 + (lane & 15);
  int g = lane >> 4;
  const _Float16* p = Sb + n * pitch + k0 + g * 8;
  v8h lo = *(const v8h*)p;
  v8h hi = *(const v8h*)(p + 16);
  return __builtin_shufflevector(lo, hi, 0,1,2,3,4,5,6,7,8,9,10,11,12,13,14,15);
}

__device__ __forceinline__ v8f wmma16(v16h a, v16h b, v8f c) {
  // (neg_a, A, neg_b, B, c_mod, C, reuse_a, reuse_b)
  return __builtin_amdgcn_wmma_f32_16x16x32_f16(false, a, false, b, (short)0, c,
                                                false, false);
}

// ---- Prepack: fp32 weights -> f16 im2col-major layout in workspace ---------
// po[32][1152]  : offset conv weights, rows 18..31 zero (k = kk*128 + c)
// pw[128][1152] : deform conv weights (k = kk*128 + c)
// p1[64][128]   : fusion 1x1 #1, p2[128][64] : fusion 1x1 #2
__global__ void dcwf_prepack(const float* __restrict__ ow,
                             const float* __restrict__ dw,
                             const float* __restrict__ w1,
                             const float* __restrict__ w2,
                             _Float16* __restrict__ po,
                             _Float16* __restrict__ pw,
                             _Float16* __restrict__ p1,
                             _Float16* __restrict__ p2) {
  int t = blockIdx.x * blockDim.x + threadIdx.x;
  if (t < 32 * KTOT) {
    int oc = t / KTOT, k = t - oc * KTOT;
    int kk = k >> 7, c = k & 127;
    float v = (oc < 18) ? ow[(oc * 128 + c) * 9 + kk] : 0.f;
    po[t] = (_Float16)v;
  }
  if (t < 128 * KTOT) {
    int co = t / KTOT, k = t - co * KTOT;
    int kk = k >> 7, c = k & 127;
    pw[t] = (_Float16)dw[(co * 128 + c) * 9 + kk];
  }
  if (t < 64 * 128) p1[t] = (_Float16)w1[t];
  if (t < 128 * 64) p2[t] = (_Float16)w2[t];
}

// ---- Main fused kernel -----------------------------------------------------
__global__ __launch_bounds__(256) void dcwf_main(
    const float* __restrict__ xin,
    const float* __restrict__ offset_b,
    const float* __restrict__ bias,
    const float* __restrict__ fus_b1,
    const float* __restrict__ fus_b2,
    const _Float16* __restrict__ po,
    const _Float16* __restrict__ pw,
    const _Float16* __restrict__ p1,
    const _Float16* __restrict__ p2,
    float* __restrict__ out) {
  extern __shared__ __align__(16) char smem[];
  _Float16* S    = (_Float16*)(smem);            // 32 x 1160 halves  (74240 B)
  _Float16* xc   = (_Float16*)(smem + 74240);    // 32 x 136  halves  ( 8704 B)
  _Float16* hbuf = (_Float16*)(smem + 82944);    // 32 x 72   halves  ( 4608 B)
  float*    offs = (float*)(smem + 87552);       // 18 x 32 f32       ( 2304 B)
  float*    tabw = (float*)(smem + 89856);       // 288 x 4 f32       ( 4608 B)
  int*      tabo = (int*)(smem + 94464);         // 288 x 4 i32       ( 4608 B)
  // total dynamic LDS: 99072 bytes

  const int tid  = threadIdx.x;
  const int lane = tid & 31;    // wave32
  const int wave = tid >> 5;    // 8 waves
  const int P0   = blockIdx.x * 32;   // 32 raster positions / WG; 6400 % 32 == 0
  const int b    = P0 / 6400;
  const int hw0  = P0 - b * 6400;
  const float* xb_base = xin + b * 128 * 6400;

  // ---- Phase 1: standard 3x3 im2col patch S0 (f16) + center slice --------
  for (int idx = tid; idx < 9 * 128 * 32; idx += 256) {
    int p   = idx & 31;
    int kkc = idx >> 5;          // kk*128 + c
    int kk  = kkc >> 7;
    int c   = kkc & 127;
    int hw  = hw0 + p;
    int h   = hw / 80, w = hw - h * 80;
    int y   = h + (kk / 3) - 1;
    int x   = w + (kk % 3) - 1;
    float v = 0.f;
    if ((unsigned)y < 80u && (unsigned)x < 80u)
      v = xb_base[(c * 80 + y) * 80 + x];
    _Float16 hv = (_Float16)v;
    S[p * SPITCH + kkc] = hv;
    if (kk == 4) xc[p * XCPITCH + c] = hv;   // center tap = x itself
  }
  __syncthreads();

  // ---- Phase 2: offset GEMM (waves 0-1) || fusion GEMM1 + silu (waves 2-5)
  if (wave < 2) {
    int m0 = wave * 16;
    v8f a0 = {0,0,0,0,0,0,0,0}, a1 = {0,0,0,0,0,0,0,0};
    for (int kc = 0; kc < KCH; ++kc) {
      int k0 = kc * 32;
      v16h af = load_a_frag(po, KTOT, m0, k0, lane);
      v16h b0 = load_b_frag(S, SPITCH, 0, k0, lane);
      v16h b1 = load_b_frag(S, SPITCH, 16, k0, lane);
      a0 = wmma16(af, b0, a0);
      a1 = wmma16(af, b1, a1);
    }
    int n = lane & 15;
    #pragma unroll
    for (int r = 0; r < 8; ++r) {
      int m = m0 + (lane >> 4) * 8 + r;
      if (m < 18) {
        float bv = offset_b[m];
        offs[m * 32 + n]      = a0[r] + bv;
        offs[m * 32 + 16 + n] = a1[r] + bv;
      }
    }
  } else if (wave < 6) {
    int m0 = (wave - 2) * 16;
    v8f a0 = {0,0,0,0,0,0,0,0}, a1 = {0,0,0,0,0,0,0,0};
    #pragma unroll
    for (int kc = 0; kc < 4; ++kc) {
      int k0 = kc * 32;
      v16h af = load_a_frag(p1, 128, m0, k0, lane);
      v16h b0 = load_b_frag(xc, XCPITCH, 0, k0, lane);
      v16h b1 = load_b_frag(xc, XCPITCH, 16, k0, lane);
      a0 = wmma16(af, b0, a0);
      a1 = wmma16(af, b1, a1);
    }
    int n = lane & 15;
    #pragma unroll
    for (int r = 0; r < 8; ++r) {
      int m = m0 + (lane >> 4) * 8 + r;
      float bv = fus_b1[m];
      float v0 = a0[r] + bv; v0 = v0 / (1.f + __expf(-v0));   // silu
      float v1 = a1[r] + bv; v1 = v1 / (1.f + __expf(-v1));
      hbuf[n * HPITCH + m]        = (_Float16)v0;   // stored [pos][ch] for B-frags
      hbuf[(16 + n) * HPITCH + m] = (_Float16)v1;
    }
  }
  __syncthreads();

  // ---- Phase 3a: bilinear corner tables (fp32) ---------------------------
  for (int e = tid; e < 288; e += 256) {    // e = kk*32 + p
    int kk = e >> 5;
    int p  = e & 31;
    int hw = hw0 + p;
    int h  = hw / 80, w = hw - h * 80;
    float ys = offs[(2 * kk) * 32 + p]     + (float)(h + kk / 3 - 1);
    float xs = offs[(2 * kk + 1) * 32 + p] + (float)(w + kk % 3 - 1);
    float y0f = floorf(ys), x0f = floorf(xs);
    int y0 = (int)y0f, x0 = (int)x0f;
    float wy1 = ys - y0f, wy0 = 1.f - wy1;
    float wx1 = xs - x0f, wx0 = 1.f - wx1;
    #pragma unroll
    for (int j = 0; j < 4; ++j) {
      int dy = j >> 1, dx = j & 1;
      int yi = y0 + dy, xi = x0 + dx;
      float wgt = (dy ? wy1 : wy0) * (dx ? wx1 : wx0);
      bool valid = (yi >= 0) && (yi < 80) && (xi >= 0) && (xi < 80);
      int yc  = min(max(yi, 0), 79);
      int xcl = min(max(xi, 0), 79);
      tabw[e * 4 + j] = valid ? wgt : 0.f;
      tabo[e * 4 + j] = yc * 80 + xcl;
    }
  }
  __syncthreads();

  // ---- Phase 3b: deformed im2col S1 (overwrites S0; gathers hit L2) ------
  for (int idx = tid; idx < 9 * 128 * 32; idx += 256) {
    int p   = idx & 31;
    int kkc = idx >> 5;
    int kk  = kkc >> 7;
    int c   = kkc & 127;
    int e   = kk * 32 + p;
    const float* xb = xb_base + c * 6400;
    float v = tabw[e * 4 + 0] * xb[tabo[e * 4 + 0]]
            + tabw[e * 4 + 1] * xb[tabo[e * 4 + 1]]
            + tabw[e * 4 + 2] * xb[tabo[e * 4 + 2]]
            + tabw[e * 4 + 3] * xb[tabo[e * 4 + 3]];
    S[p * SPITCH + kkc] = (_Float16)v;
  }
  __syncthreads();

  // ---- Phase 4: main deform GEMM, M=128 (wave -> 16 rows), N=32, K=1152 --
  const int m0 = wave * 16;
  const _Float16* pwm = pw + (m0 + (lane & 15)) * KTOT;   // this lane's A row
  v8f acc0 = {0,0,0,0,0,0,0,0}, acc1 = {0,0,0,0,0,0,0,0};
  for (int kc = 0; kc < KCH; ++kc) {
    int k0 = kc * 32;
    if (kc + 1 < KCH)
      __builtin_prefetch((const void*)(pwm + k0 + 32), 0, 1);  // global_prefetch_b8
    v16h af = load_a_frag(pw, KTOT, m0, k0, lane);
    v16h b0 = load_b_frag(S, SPITCH, 0, k0, lane);
    v16h b1 = load_b_frag(S, SPITCH, 16, k0, lane);
    acc0 = wmma16(af, b0, acc0);
    acc1 = wmma16(af, b1, acc1);
  }

  // ---- Phase 5: fusion GEMM2 accumulated into the same accumulators ------
  #pragma unroll
  for (int kc = 0; kc < 2; ++kc) {
    int k0 = kc * 32;
    v16h af = load_a_frag(p2, 64, m0, k0, lane);
    v16h b0 = load_b_frag(hbuf, HPITCH, 0, k0, lane);
    v16h b1 = load_b_frag(hbuf, HPITCH, 16, k0, lane);
    acc0 = wmma16(af, b0, acc0);
    acc1 = wmma16(af, b1, acc1);
  }

  // ---- Phase 6: bias + store (NCHW, 32 contiguous positions) -------------
  int n = lane & 15;
  float* ob = out + (b * 128) * 6400 + hw0;
  #pragma unroll
  for (int r = 0; r < 8; ++r) {
    int m = m0 + (lane >> 4) * 8 + r;
    float bsum = bias[m] + fus_b2[m];
    ob[m * 6400 + n]      = acc0[r] + bsum;
    ob[m * 6400 + 16 + n] = acc1[r] + bsum;
  }
}

// ---------------------------------------------------------------------------
extern "C" void kernel_launch(void* const* d_in, const int* in_sizes, int n_in,
                              void* d_out, int out_size, void* d_ws, size_t ws_size,
                              hipStream_t stream) {
  (void)in_sizes; (void)n_in; (void)out_size; (void)ws_size;
  const float* x        = (const float*)d_in[0];
  const float* offset_w = (const float*)d_in[1];
  const float* offset_b = (const float*)d_in[2];
  const float* weight   = (const float*)d_in[3];
  const float* bias     = (const float*)d_in[4];
  const float* fus_w1   = (const float*)d_in[5];
  const float* fus_b1   = (const float*)d_in[6];
  const float* fus_w2   = (const float*)d_in[7];
  const float* fus_b2   = (const float*)d_in[8];
  float* out = (float*)d_out;

  // workspace carve (f16 prepacked weights), ~392 KB total
  char* ws = (char*)d_ws;
  _Float16* po = (_Float16*)(ws + 0);                         // 32*1152
  _Float16* pw = (_Float16*)(ws + 73728);                     // 128*1152
  _Float16* p1 = (_Float16*)(ws + 73728 + 294912);            // 64*128
  _Float16* p2 = (_Float16*)(ws + 73728 + 294912 + 16384);    // 128*64

  dcwf_prepack<<<576, 256, 0, stream>>>(offset_w, weight, fus_w1, fus_w2,
                                        po, pw, p1, p2);

  const int SMEM = 99072;   // > 64 KB dynamic LDS: opt in (idempotent, capture-safe)
  (void)hipFuncSetAttribute((const void*)dcwf_main,
                            hipFuncAttributeMaxDynamicSharedMemorySize, SMEM);
  dcwf_main<<<1600, 256, SMEM, stream>>>(x, offset_b, bias, fus_b1, fus_b2,
                                         po, pw, p1, p2, out);
}
